// WeatherGNNEncoder_49254684950776
// MI455X (gfx1250) — matrix operand order repaired and checked by reading, not compiled, not run on hardware
//
#include <hip/hip_runtime.h>
#include <math.h>

// ---------------- problem constants ----------------
#define H_GRID 361
#define W_GRID 720
#define HW_GRID 259920
#define C_IN_ 78
#define N_NODES_ 5888
#define K_NB 64
#define D_HID_ 256
#define RADIUS_KM 82.5f
#define EARTH_R 6371.0f
#define DEG2RAD 0.017453292519943295f

#define X_STRIDE 264   // halfs per row of f16 activation tile (8-half pad)
#define W_STRIDE 136   // halfs per row of f16 weight tile (8-half pad, matches TDM pad)
#define W_HALF   (256 * W_STRIDE)   // halfs per staged K-half

typedef __attribute__((ext_vector_type(16))) _Float16 v16h;
typedef __attribute__((ext_vector_type(8)))  _Float16 v8h;
typedef __attribute__((ext_vector_type(8)))  float    v8f;
typedef unsigned int u32x4 __attribute__((ext_vector_type(4)));
typedef int          i32x8 __attribute__((ext_vector_type(8)));
typedef int          i32x4 __attribute__((ext_vector_type(4)));

#if defined(__has_builtin)
#if __has_builtin(__builtin_amdgcn_tensor_load_to_lds) && __has_builtin(__builtin_amdgcn_s_wait_tensorcnt)
#define HAVE_TDM 1
#else
#define HAVE_TDM 0
#endif
#else
#define HAVE_TDM 0
#endif

static __device__ __forceinline__ v16h cat16(v8h lo, v8h hi) {
    return __builtin_shufflevector(lo, hi, 0,1,2,3,4,5,6,7,8,9,10,11,12,13,14,15);
}

// =====================================================================
// Kernel 1: geometry. One block (128 thr) per node. Only points within
// 82.5 km can affect any output; they all lie in a 5-row x 25-col window
// (wrapping in lon). Collect, rank-sort (matches top_k tie-break), emit
// idx / normalized weights / coords / valid.
// =====================================================================
__global__ void geom_kernel(const float* __restrict__ ico,
                            int*   __restrict__ idx,
                            float* __restrict__ wgt,
                            float* __restrict__ coords,
                            float* __restrict__ validf) {
    __shared__ float sd[K_NB], slat[K_NB], slon[K_NB];
    __shared__ int   sidx[K_NB];
    __shared__ float sd2[K_NB], slat2[K_NB], slon2[K_NB];
    __shared__ int   sidx2[K_NB];
    __shared__ int   scnt;
    __shared__ float swsum;

    const int n = blockIdx.x;
    const int t = threadIdx.x;
    if (t == 0) scnt = 0;
    __syncthreads();

    const float lat_deg = ico[2*n];
    const float lon_deg = ico[2*n+1];
    const float lat_r = lat_deg * DEG2RAD;
    const float lon_r = lon_deg * DEG2RAD;
    const int ilat = (int)roundf((lat_deg + 90.0f) * 2.0f);
    const int ilon = (int)roundf(lon_deg * (719.0f / 360.0f));

    if (t < 125) {
        const int dr = t / 25 - 2;
        const int dc = t % 25 - 12;
        const int r = ilat + dr;
        if (r >= 0 && r < H_GRID) {
            int c = ilon + dc;
            c = ((c % W_GRID) + W_GRID) % W_GRID;
            const float latg = (-90.0f + 0.5f * (float)r) * DEG2RAD;
            const float long_ = ((float)c * (360.0f / 719.0f)) * DEG2RAD;
            const float sdl = sinf((latg - lat_r) * 0.5f);
            const float sdo = sinf((long_ - lon_r) * 0.5f);
            float a = sdl*sdl + cosf(lat_r) * cosf(latg) * sdo*sdo;
            a = fminf(fmaxf(a, 0.0f), 1.0f);
            const float d = 2.0f * asinf(sqrtf(a)) * EARTH_R;
            if (d <= RADIUS_KM) {
                const int p = atomicAdd(&scnt, 1);
                if (p < K_NB) { sd[p]=d; sidx[p]=r*W_GRID+c; slat[p]=latg; slon[p]=long_; }
            }
        }
    }
    __syncthreads();
    const int cnt = min(scnt, K_NB);
    if (t < cnt) {  // O(n^2) rank sort; tie-break on lower flat index (matches top_k)
        const float d = sd[t]; const int id = sidx[t];
        int rank = 0;
        for (int j = 0; j < cnt; ++j) {
            const float dj = sd[j]; const int ij = sidx[j];
            rank += (dj < d) || (dj == d && ij < id);
        }
        sd2[rank]=d; sidx2[rank]=id; slat2[rank]=slat[t]; slon2[rank]=slon[t];
    }
    __syncthreads();
    if (t == 0) {
        float s = 0.0f;
        for (int j = 0; j < cnt; ++j) s += expf(-sd2[j] / RADIUS_KM);
        swsum = (s > 0.0f) ? s : 1.0f;
    }
    __syncthreads();
    if (t < K_NB) {
        const size_t base = (size_t)n * K_NB + t;
        float* cp = coords + base * 5;
        if (t < cnt) {
            idx[base]    = sidx2[t];
            wgt[base]    = expf(-sd2[t] / RADIUS_KM) / swsum;
            validf[base] = 1.0f;
            const float latp = slat2[t], lonp = slon2[t];
            cp[0] = latp - lat_deg;            // faithful rad-minus-deg
            cp[1] = lonp - lon_deg;
            cp[2] = sd2[t];
            cp[3] = cosf(latp - lat_r);
            cp[4] = sinf(lonp - lon_r);
        } else {
            idx[base] = 0; wgt[base] = 0.0f; validf[base] = 0.0f;
            cp[0]=cp[1]=cp[2]=cp[3]=cp[4]=0.0f;
        }
    }
}

// =====================================================================
// Kernel 2: transpose + f16-convert the 256x256 layer-2 weights so B
// fragments are contiguous 32B runs and TDM can fetch row-major tiles.
// =====================================================================
__global__ void wt_kernel(const float* __restrict__ ew2, const float* __restrict__ nw2,
                          _Float16* __restrict__ ewt, _Float16* __restrict__ nwt) {
    const int t = blockIdx.x * 256 + threadIdx.x;
    if (t < 65536) {
        const int n = t >> 8, k = t & 255;
        ewt[t] = (_Float16)ew2[k * 256 + n];
    } else {
        const int t2 = t - 65536;
        const int n = t2 >> 8, k = t2 & 255;
        nwt[t2] = (_Float16)nw2[k * 256 + n];
    }
}

// =====================================================================
// Kernel 3: weighted neighbor aggregation. One wave per node; lane = 2
// neighbor slots; xor-shuffle reduce. Neighbor indices are spatially
// adjacent -> mostly-coalesced gathers.
// =====================================================================
__global__ void agg_kernel(const float* __restrict__ data,
                           const float* __restrict__ means,
                           const float* __restrict__ stds,
                           const int*   __restrict__ idx,
                           const float* __restrict__ wgt,
                           float* __restrict__ agg) {
    const int lane = threadIdx.x & 31;
    const int wid  = __builtin_amdgcn_readfirstlane(threadIdx.x >> 5);
    const int n = blockIdx.x * 8 + wid;
    const size_t b = (size_t)n * K_NB;
    const int   i0 = idx[b + lane],      i1 = idx[b + lane + 32];
    const float w0 = wgt[b + lane],      w1 = wgt[b + lane + 32];
    for (int c = 0; c < C_IN_; ++c) {
        const float mu  = means[c];
        const float inv = 1.0f / (stds[c] + 1e-7f);
        const float* dc = data + (size_t)c * HW_GRID;
        float v = w0 * (dc[i0] - mu) * inv + w1 * (dc[i1] - mu) * inv;
        #pragma unroll
        for (int m = 16; m >= 1; m >>= 1) v += __shfl_xor(v, m, 32);
        if (lane == 0) agg[(size_t)n * 80 + c] = v;
    }
}

// =====================================================================
// TDM: one-instruction DMA of a 128-col x 256-row f16 tile (one K-half of
// W2^T) from global into LDS, with descriptor padding of 4 dwords every
// 64 dwords -> LDS row stride of 136 halfs (bank-conflict padding) free.
// =====================================================================
static __device__ __forceinline__ void tdm_load_half(const _Float16* __restrict__ gsrc,
                                                     unsigned lds_byte_addr) {
#if HAVE_TDM
    const unsigned long long ga = (unsigned long long)(uintptr_t)gsrc;
    u32x4 g0 = {};
    g0[0] = 1u;                                            // count=1 (valid user D#)
    g0[1] = lds_byte_addr;                                 // lds_addr
    g0[2] = (unsigned)(ga & 0xffffffffu);                  // global_addr[31:0]
    g0[3] = (unsigned)((ga >> 32) & 0x01ffffffu) | (2u << 30);  // addr[56:32] | type=2
    i32x8 g1 = {};
    g1[0] = (1 << 16)      // data_size = 2 bytes
          | (1 << 20)      // pad_enable
          | (5 << 22)      // pad_interval: every 64 dwords (=128 halfs = one row)
          | (3 << 25);     // pad_amount: 4 dwords (=8 halfs)
    g1[1] = (int)(256u << 16);   // tensor_dim0 = 256 (full row length)
    g1[2] = (int)(256u << 16);   // tensor_dim1 = 256 rows
    g1[3] = (int)(128u << 16);   // tile_dim0 = 128 (one K-half)
    g1[4] = 256;                 // tile_dim1 = 256 rows
    g1[5] = 256;                 // tensor_dim0_stride = 256
    i32x4 gz = {};
#if __clang_major__ >= 23
    i32x8 gz8 = {};
    __builtin_amdgcn_tensor_load_to_lds(g0, g1, gz, gz, gz8, 0);
#else
    __builtin_amdgcn_tensor_load_to_lds(g0, g1, gz, gz, 0);
#endif
#else
    (void)gsrc; (void)lds_byte_addr;
#endif
}

// =====================================================================
// Kernel 4: fused 2-layer MLP (relu->LN, relu->LN) with WMMA layer-2.
// Block = 256 thr (8 waves); tile = 128 rows x 256 cols.
//   entry  : wave 0 issues TWO tensor_load_to_lds DMAs (both K-halves of
//            W2^T) -> DMA overlaps the whole layer-1 phase
//   stage 1: h1 = LN(relu(X1 @ W1 + b1)) -> f16 LDS tile (wave-per-row LN)
//   stage 2: s_wait_tensorcnt(0) + barrier, then 128 straight WMMAs with
//            double-buffered B fragments from LDS (partial dscnt waits)
//   stage 3: bias+relu, in-register LN via xor-shuffle over the D-matrix
//            lane layout, optional per-row mask (edges), store f32.
// =====================================================================
template<int K1, bool MASK>
__global__ __launch_bounds__(256) void mlp_kernel(
        const float* __restrict__ X1, int s1,
        const float* __restrict__ W1, const float* __restrict__ b1,
        const float* __restrict__ g1v, const float* __restrict__ o1,
        const _Float16* __restrict__ Wt,
        const float* __restrict__ b2, const float* __restrict__ g2,
        const float* __restrict__ o2,
        const float* __restrict__ maskw,
        float* __restrict__ out) {
    __shared__ __align__(16) _Float16 Xl[128 * X_STRIDE];   //  67,584 B
    __shared__ __align__(16) _Float16 Wl[2 * W_HALF];       // 139,264 B

    const int lane = threadIdx.x & 31;
    const int wid  = __builtin_amdgcn_readfirstlane(threadIdx.x >> 5);
    const size_t m0 = (size_t)blockIdx.x * 128;

#if HAVE_TDM
    // issue both W2^T K-half DMAs up front; they overlap all of stage 1
    if (threadIdx.x < 32) {
        tdm_load_half(Wt,       (unsigned)(uintptr_t)(&Wl[0]));
        tdm_load_half(Wt + 128, (unsigned)(uintptr_t)(&Wl[W_HALF]));
    }
#endif

    // ---- stage 1: layer 1 + LN, wave handles rows [16*wid, 16*wid+16)
    for (int i = 0; i < 16; ++i) {
        const int lr = wid * 16 + i;
        const float* xr = X1 + (m0 + lr) * (size_t)s1;
        float v[8];
        float s = 0.0f, sq = 0.0f;
        #pragma unroll
        for (int j = 0; j < 8; ++j) {
            const int c = lane + 32 * j;
            float a = b1[c];
            for (int k = 0; k < K1; ++k) a = fmaf(xr[k], W1[k * 256 + c], a);
            a = fmaxf(a, 0.0f);
            v[j] = a; s += a; sq += a * a;
        }
        #pragma unroll
        for (int m = 16; m >= 1; m >>= 1) { s += __shfl_xor(s, m, 32); sq += __shfl_xor(sq, m, 32); }
        const float mean = s * (1.0f / 256.0f);
        const float var  = sq * (1.0f / 256.0f) - mean * mean;
        const float rs   = rsqrtf(var + 1e-5f);
        #pragma unroll
        for (int j = 0; j < 8; ++j) {
            const int c = lane + 32 * j;
            Xl[lr * X_STRIDE + c] = (_Float16)((v[j] - mean) * rs * g1v[c] + o1[c]);
        }
    }

#if HAVE_TDM
    if (threadIdx.x < 32) __builtin_amdgcn_s_wait_tensorcnt(0);
#else
    for (int i = threadIdx.x; i < 8192; i += 256) {  // fallback: cooperative copy
        const int kh = i >> 12, j = i & 4095;
        const int n = j >> 4, kc = j & 15;
        *reinterpret_cast<v8h*>(&Wl[kh * W_HALF + n * W_STRIDE + kc * 8]) =
            *reinterpret_cast<const v8h*>(&Wt[n * 256 + kh * 128 + kc * 8]);
    }
#endif
    __syncthreads();

    // ---- stage 2: 128 WMMAs, both K-halves resident, B double-buffered
    const int half = lane >> 4;         // 0: lanes 0-15, 1: lanes 16-31
    const int l15  = lane & 15;
    const int arow = wid * 16 + l15;
    v8f acc[16] = {};
    for (int kh = 0; kh < 2; ++kh) {
        const _Float16* wl = &Wl[kh * W_HALF];
        for (int ks = 0; ks < 4; ++ks) {
            // A fragment (contiguous K, matching B): lanes<16 K{0..15}, lanes>=16 K{16..31}
            const int ka = kh * 128 + ks * 32 + half * 16;
            const v8h alo = *reinterpret_cast<const v8h*>(&Xl[arow * X_STRIDE + ka]);
            const v8h ahi = *reinterpret_cast<const v8h*>(&Xl[arow * X_STRIDE + ka + 8]);
            const v16h a = cat16(alo, ahi);
            const int kb = ks * 32 + half * 16;   // within the staged half
            v16h bf[2];
            bf[0] = cat16(*reinterpret_cast<const v8h*>(&wl[l15 * W_STRIDE + kb]),
                          *reinterpret_cast<const v8h*>(&wl[l15 * W_STRIDE + kb + 8]));
            #pragma unroll
            for (int nt = 0; nt < 16; ++nt) {
                if (nt < 15) {   // prefetch next B fragment into the other buffer
                    const int ncol = (nt + 1) * 16 + l15;
                    bf[(nt + 1) & 1] =
                        cat16(*reinterpret_cast<const v8h*>(&wl[ncol * W_STRIDE + kb]),
                              *reinterpret_cast<const v8h*>(&wl[ncol * W_STRIDE + kb + 8]));
                }
                acc[nt] = __builtin_amdgcn_wmma_f32_16x16x32_f16(
                    false, a, false, bf[nt & 1], (short)0, acc[nt], false, false);
            }
        }
    }

    // ---- stage 3: bias + relu, LN per row fully in registers, mask, store
    #pragma unroll
    for (int nt = 0; nt < 16; ++nt) {
        const float bb = b2[nt * 16 + l15];
        #pragma unroll
        for (int v = 0; v < 8; ++v) acc[nt][v] = fmaxf(acc[nt][v] + bb, 0.0f);
    }
    #pragma unroll
    for (int v = 0; v < 8; ++v) {
        float s = 0.0f, q = 0.0f;
        #pragma unroll
        for (int nt = 0; nt < 16; ++nt) { const float x = acc[nt][v]; s += x; q += x * x; }
        #pragma unroll
        for (int m = 8; m >= 1; m >>= 1) { s += __shfl_xor(s, m, 32); q += __shfl_xor(q, m, 32); }
        const float mean = s * (1.0f / 256.0f);
        const float var  = q * (1.0f / 256.0f) - mean * mean;
        const float rs   = rsqrtf(var + 1e-5f);
        const size_t row = m0 + (size_t)(wid * 16 + v + half * 8);
        float mk = 1.0f;
        if (MASK) mk = (maskw[row] > 0.0f) ? 1.0f : 0.0f;
        #pragma unroll
        for (int nt = 0; nt < 16; ++nt) {
            const int c = nt * 16 + l15;
            out[row * 256 + c] = ((acc[nt][v] - mean) * rs * g2[c] + o2[c]) * mk;
        }
    }
}

// =====================================================================
extern "C" void kernel_launch(void* const* d_in, const int* in_sizes, int n_in,
                              void* d_out, int out_size, void* d_ws, size_t ws_size,
                              hipStream_t stream) {
    (void)in_sizes; (void)n_in; (void)out_size; (void)ws_size;
    const float* data  = (const float*)d_in[0];
    const float* means = (const float*)d_in[1];
    const float* stds  = (const float*)d_in[2];
    const float* ew1 = (const float*)d_in[3];
    const float* eb1 = (const float*)d_in[4];
    const float* eg1 = (const float*)d_in[5];
    const float* eo1 = (const float*)d_in[6];
    const float* ew2 = (const float*)d_in[7];
    const float* eb2 = (const float*)d_in[8];
    const float* eg2 = (const float*)d_in[9];
    const float* eo2 = (const float*)d_in[10];
    const float* nw1 = (const float*)d_in[11];
    const float* nb1 = (const float*)d_in[12];
    const float* ng1 = (const float*)d_in[13];
    const float* no1 = (const float*)d_in[14];
    const float* nw2 = (const float*)d_in[15];
    const float* nb2 = (const float*)d_in[16];
    const float* ng2 = (const float*)d_in[17];
    const float* no2 = (const float*)d_in[18];
    const float* ico = (const float*)d_in[19];

    char* ws = (char*)d_ws;
    size_t off = 0;
    auto alloc = [&](size_t bytes) { void* p = ws + off; off = (off + bytes + 255) & ~(size_t)255; return p; };
    int*      idx    = (int*)     alloc((size_t)N_NODES_ * K_NB * 4);
    float*    wgt    = (float*)   alloc((size_t)N_NODES_ * K_NB * 4);
    float*    coords = (float*)   alloc((size_t)N_NODES_ * K_NB * 5 * 4);
    float*    agg    = (float*)   alloc((size_t)N_NODES_ * 80 * 4);
    _Float16* ewt    = (_Float16*)alloc((size_t)65536 * 2);
    _Float16* nwt    = (_Float16*)alloc((size_t)65536 * 2);

    float* out_nodes = (float*)d_out;
    float* out_edges = out_nodes + (size_t)N_NODES_ * D_HID_;
    float* out_valid = out_edges + (size_t)N_NODES_ * K_NB * D_HID_;

    geom_kernel<<<N_NODES_, 128, 0, stream>>>(ico, idx, wgt, coords, out_valid);
    wt_kernel<<<512, 256, 0, stream>>>(ew2, nw2, ewt, nwt);
    agg_kernel<<<N_NODES_ / 8, 256, 0, stream>>>(data, means, stds, idx, wgt, agg);
    mlp_kernel<C_IN_, false><<<N_NODES_ / 128, 256, 0, stream>>>(
        agg, 80, nw1, nb1, ng1, no1, nwt, nb2, ng2, no2, nullptr, out_nodes);
    mlp_kernel<5, true><<<(N_NODES_ * K_NB) / 128, 256, 0, stream>>>(
        coords, 5, ew1, eb1, eg1, eo1, ewt, eb2, eg2, eo2, wgt, out_edges);
}
